// WrapperModel_83966610637255
// MI455X (gfx1250) — matrix operand order repaired and checked by reading, not compiled, not run on hardware
//
#include <hip/hip_runtime.h>
#include <math.h>

// ---- model dims (match reference) ----
#define S_   2048
#define D_   1024
#define H_   16
#define DH_  64
#define L_   2
#define E_   8
#define I_   512
#define SI_  1024
#define V_   32000
#define IGNORE_IDX (-100)

typedef __attribute__((ext_vector_type(16))) __bf16 v16bf;
typedef __attribute__((ext_vector_type(8)))  __bf16 bf16x8;
typedef __attribute__((ext_vector_type(4)))  __bf16 bf16x4;
typedef __attribute__((ext_vector_type(8)))  float  v8f;

__device__ __forceinline__ v8f wmma_bf16(v16bf a, v16bf b, v8f c) {
  // (neg_a, A, neg_b, B, c_mod, C, reuse_a, reuse_b)
  return __builtin_amdgcn_wmma_f32_16x16x32_bf16(false, a, false, b, (short)0, c,
                                                 false, false);
}

// ---- WMMA fragment loaders (ISA 7.12.2 layouts, wave32, bf16 sources) ----
// A fragment: 16x32 tile, row-major bf16. lane m=lane&15 holds
// K in [8*hi, 8*hi+8) and [8*hi+16, 8*hi+24), hi = lane>>4.  Two b128 loads.
__device__ __forceinline__ v16bf frag_a_bf(const __bf16* tile, int ld) {
  int lane = threadIdx.x & 31;
  const __bf16* r = tile + (lane & 15) * ld + ((lane >> 4) << 3);
  bf16x8 lo = *(const bf16x8*)r;
  bf16x8 hi = *(const bf16x8*)(r + 16);
  return __builtin_shufflevector(lo, hi, 0, 1, 2, 3, 4, 5, 6, 7,
                                 8, 9, 10, 11, 12, 13, 14, 15);
}

// B fragment where B[k][n] = W[n][k] (C = A @ W^T, W row-major NxK bf16).
// lane n=lane&15 holds K in [8*hi, 8*hi+8) and [8*hi+16, 8*hi+24), hi = lane>>4.
__device__ __forceinline__ v16bf frag_b_wt_bf(const __bf16* w, int ld) {
  int lane = threadIdx.x & 31;
  const __bf16* r = w + (lane & 15) * ld + ((lane >> 4) << 3);
  bf16x8 lo = *(const bf16x8*)r;
  bf16x8 hi = *(const bf16x8*)(r + 16);
  return __builtin_shufflevector(lo, hi, 0, 1, 2, 3, 4, 5, 6, 7,
                                 8, 9, 10, 11, 12, 13, 14, 15);
}

// C/D fragment store: lane holds col n=lane&15, rows rb+r, rb = 8*(lane>>4).
template <bool ADD>
__device__ __forceinline__ void store_c(float* tile, int ld, v8f c) {
  int lane = threadIdx.x & 31;
  int n = lane & 15, rb = (lane >> 4) << 3;
#pragma unroll
  for (int r = 0; r < 8; ++r) {
    float* p = tile + (rb + r) * ld + n;
    if (ADD) *p += c[r]; else *p = c[r];
  }
}

// ---- GEMM: C[M,N] (= or +=) A[M,K] @ W[N,K]^T ; one wave per 32x64 tile ----
template <bool ADD>
__global__ __launch_bounds__(32)
void gemm_bf(const __bf16* __restrict__ A, const __bf16* __restrict__ W,
             float* __restrict__ C, int Kd, int lda, int ldw, int ldc) {
  int m0 = blockIdx.y << 5;
  int n0 = blockIdx.x << 6;
  const __bf16* a0 = A + (size_t)m0 * lda;
  const __bf16* a1 = a0 + (size_t)16 * lda;
  const __bf16* w0 = W + (size_t)n0 * ldw;
  v8f acc[2][4] = {};
  for (int kc = 0; kc < Kd; kc += 32) {
    __builtin_prefetch(a0 + kc + 32, 0, 3);          // global_prefetch_b8
    __builtin_prefetch(w0 + kc + 32, 0, 3);
    v16bf fa0 = frag_a_bf(a0 + kc, lda);
    v16bf fa1 = frag_a_bf(a1 + kc, lda);
#pragma unroll
    for (int t = 0; t < 4; ++t) {
      v16bf fb = frag_b_wt_bf(w0 + (size_t)(t << 4) * ldw + kc, ldw);
      acc[0][t] = wmma_bf16(fa0, fb, acc[0][t]);
      acc[1][t] = wmma_bf16(fa1, fb, acc[1][t]);
    }
  }
  float* ct = C + (size_t)m0 * ldc + n0;
#pragma unroll
  for (int t = 0; t < 4; ++t) {
    store_c<ADD>(ct + (t << 4), ldc, acc[0][t]);
    store_c<ADD>(ct + (size_t)16 * ldc + (t << 4), ldc, acc[1][t]);
  }
}

// ---- conversion kernels (hardware f32<->bf16 casts, vectorized) ----
__global__ void cvt_bf16_kernel(const float* __restrict__ src, __bf16* __restrict__ dst) {
  int idx = (blockIdx.x * 256 + threadIdx.x) << 2;
  float4 v = *(const float4*)(src + idx);
  bf16x4 o = { (__bf16)v.x, (__bf16)v.y, (__bf16)v.z, (__bf16)v.w };
  *(bf16x4*)(dst + idx) = o;
}

// v [S,D] f32 -> vt [D,S] bf16 (transpose so attention P@V B-frags are contiguous)
__global__ void cvt_vT_kernel(const float* __restrict__ v, __bf16* __restrict__ vt) {
  int idx = blockIdx.x * 256 + threadIdx.x;   // idx = c*S + s
  int s = idx & (S_ - 1);
  int c = idx >> 11;
  vt[idx] = (__bf16)v[(size_t)s * D_ + c];
}

// ---- elementwise / small kernels ----
__global__ void embed_kernel(const int* __restrict__ toks,
                             const float* __restrict__ emb, float* __restrict__ h) {
  int idx = blockIdx.x * 256 + threadIdx.x;
  int s = idx >> 10, d = idx & (D_ - 1);
  h[idx] = emb[(size_t)toks[s] * D_ + d];
}

__global__ void rmsnorm_kernel(const float* __restrict__ X,
                               const float* __restrict__ w, __bf16* __restrict__ Y) {
  __shared__ float red[256];
  int s = blockIdx.x;
  const float* x = X + (size_t)s * D_;
  float ss = 0.f;
  for (int d = threadIdx.x; d < D_; d += 256) { float t = x[d]; ss += t * t; }
  red[threadIdx.x] = ss;
  __syncthreads();
  for (int st = 128; st > 0; st >>= 1) {
    if (threadIdx.x < st) red[threadIdx.x] += red[threadIdx.x + st];
    __syncthreads();
  }
  float inv = rsqrtf(red[0] / (float)D_ + 1e-6f);
  for (int d = threadIdx.x; d < D_; d += 256)
    Y[(size_t)s * D_ + d] = (__bf16)(x[d] * inv * w[d]);
}

// rotary: q,k f32 in -> qb,kb bf16 out; folds 1/sqrt(Dh) into q
__global__ void rope_kernel(const float* __restrict__ q, const float* __restrict__ k,
                            __bf16* __restrict__ qb, __bf16* __restrict__ kb) {
  int idx = blockIdx.x * 256 + threadIdx.x;   // S*H*(DH/2)
  int i = idx & 31;
  int hh = (idx >> 5) & (H_ - 1);
  int s = idx >> 9;
  float freq = __powf(10000.f, -(float)(2 * i) / (float)DH_);
  float ang = (float)s * freq, sn, cs;
  __sincosf(ang, &sn, &cs);
  size_t b = (size_t)s * D_ + hh * DH_ + i;
  const float scale = 0.125f;                 // 1/sqrt(64)
  float q1 = q[b], q2 = q[b + 32];
  qb[b]      = (__bf16)((q1 * cs - q2 * sn) * scale);
  qb[b + 32] = (__bf16)((q2 * cs + q1 * sn) * scale);
  float k1 = k[b], k2 = k[b + 32];
  kb[b]      = (__bf16)(k1 * cs - k2 * sn);
  kb[b + 32] = (__bf16)(k2 * cs + k1 * sn);
}

// ---- flash attention: one wave per (16-query tile, head); all-bf16 operands ----
__global__ __launch_bounds__(32)
void attn_kernel(const __bf16* __restrict__ Q, const __bf16* __restrict__ K,
                 const __bf16* __restrict__ VT, __bf16* __restrict__ O) {
  __shared__ __bf16 pls[16 * 32];
  int q0 = blockIdx.x << 4;
  int hh = blockIdx.y;
  int lane = threadIdx.x;
  int n = lane & 15, rb = (lane >> 4) << 3;
  const __bf16* qp = Q + (size_t)q0 * D_ + hh * DH_;
  v16bf aq0 = frag_a_bf(qp, D_);
  v16bf aq1 = frag_a_bf(qp + 32, D_);
  v8f o0 = {}, o1 = {}, o2 = {}, o3 = {};
  float m[8], l[8];
#pragma unroll
  for (int r = 0; r < 8; ++r) { m[r] = -1e30f; l[r] = 0.f; }
  int nk = (q0 >> 5) + 1;
  for (int kb = 0; kb < nk; ++kb) {
    int kc = kb << 5;
    const __bf16* kp = K + (size_t)kc * D_ + hh * DH_;
    v8f s0 = {}, s1 = {};
    s0 = wmma_bf16(aq0, frag_b_wt_bf(kp, D_), s0);
    s0 = wmma_bf16(aq1, frag_b_wt_bf(kp + 32, D_), s0);
    s1 = wmma_bf16(aq0, frag_b_wt_bf(kp + 16 * D_, D_), s1);
    s1 = wmma_bf16(aq1, frag_b_wt_bf(kp + 16 * D_ + 32, D_), s1);
#pragma unroll
    for (int r = 0; r < 8; ++r) {
      int row = q0 + rb + r;
      if (kc + n > row)      s0[r] = -1e30f;
      if (kc + 16 + n > row) s1[r] = -1e30f;
      float mx = fmaxf(s0[r], s1[r]);
      for (int d = 1; d < 16; d <<= 1) mx = fmaxf(mx, __shfl_xor(mx, d, 32));
      mx = fmaxf(mx, m[r]);
      float alpha = __expf(m[r] - mx);
      float p0 = __expf(s0[r] - mx);
      float p1 = __expf(s1[r] - mx);
      float rs = p0 + p1;
      for (int d = 1; d < 16; d <<= 1) rs += __shfl_xor(rs, d, 32);
      l[r] = l[r] * alpha + rs;
      m[r] = mx;
      o0[r] *= alpha; o1[r] *= alpha; o2[r] *= alpha; o3[r] *= alpha;
      pls[(rb + r) * 32 + n]      = (__bf16)p0;
      pls[(rb + r) * 32 + 16 + n] = (__bf16)p1;
    }
    __syncthreads();
    v16bf ap = frag_a_bf(pls, 32);
    // VT is [D][S]: B[k][n] = V[kc+k][c0+n] = VT[(c0+n)*S + kc+k]
    const __bf16* vp = VT + (size_t)(hh * DH_) * S_ + kc;
    o0 = wmma_bf16(ap, frag_b_wt_bf(vp,                    S_), o0);
    o1 = wmma_bf16(ap, frag_b_wt_bf(vp + (size_t)16 * S_,  S_), o1);
    o2 = wmma_bf16(ap, frag_b_wt_bf(vp + (size_t)32 * S_,  S_), o2);
    o3 = wmma_bf16(ap, frag_b_wt_bf(vp + (size_t)48 * S_,  S_), o3);
    __syncthreads();
  }
  __bf16* op = O + (size_t)q0 * D_ + hh * DH_;
#pragma unroll
  for (int r = 0; r < 8; ++r) {
    float inv = 1.f / l[r];
    size_t ro = (size_t)(rb + r) * D_;
    op[ro + n]      = (__bf16)(o0[r] * inv);
    op[ro + 16 + n] = (__bf16)(o1[r] * inv);
    op[ro + 32 + n] = (__bf16)(o2[r] * inv);
    op[ro + 48 + n] = (__bf16)(o3[r] * inv);
  }
}

// ---- MoE gate: softmax over E=8, top-2 combine weights; one wave per token ----
__global__ __launch_bounds__(32)
void gate_kernel(const __bf16* __restrict__ X, const float* __restrict__ GW,
                 float* __restrict__ comb) {
  int s = blockIdx.x, lane = threadIdx.x;
  const __bf16* x = X + (size_t)s * D_;
  float lg[E_];
#pragma unroll
  for (int e = 0; e < E_; ++e) {
    float p = 0.f;
    for (int d = lane; d < D_; d += 32) p += (float)x[d] * GW[(size_t)e * D_ + d];
    for (int dd = 1; dd < 32; dd <<= 1) p += __shfl_xor(p, dd, 32);
    lg[e] = p;
  }
  float mx = lg[0];
#pragma unroll
  for (int e = 1; e < E_; ++e) mx = fmaxf(mx, lg[e]);
  float se = 0.f;
#pragma unroll
  for (int e = 0; e < E_; ++e) { lg[e] = __expf(lg[e] - mx); se += lg[e]; }
#pragma unroll
  for (int e = 0; e < E_; ++e) lg[e] /= se;
  int i1 = 0;
#pragma unroll
  for (int e = 1; e < E_; ++e) if (lg[e] > lg[i1]) i1 = e;
  int i2 = (i1 == 0) ? 1 : 0;
#pragma unroll
  for (int e = 0; e < E_; ++e) if (e != i1 && lg[e] > lg[i2]) i2 = e;
  if (lane < E_)
    comb[(size_t)s * E_ + lane] = (lane == i1) ? lg[i1] : (lane == i2) ? lg[i2] : 0.f;
}

__global__ void moe_act_kernel(const float* __restrict__ g, const float* __restrict__ u,
                               const float* __restrict__ comb, __bf16* __restrict__ gb) {
  int idx = blockIdx.x * 256 + threadIdx.x;   // S*E*I
  int se = idx / I_;                          // s*E + e
  float gv = g[idx];
  float silu = gv / (1.f + __expf(-gv));
  gb[idx] = (__bf16)(silu * u[idx] * comb[se]);
}

__global__ void silu_mul_kernel(const float* __restrict__ a, const float* __restrict__ b,
                                __bf16* __restrict__ o) {
  int idx = blockIdx.x * 256 + threadIdx.x;
  float x = a[idx];
  o[idx] = (__bf16)((x / (1.f + __expf(-x))) * b[idx]);
}

__global__ void zero_kernel(float* out) { out[0] = 0.f; }

// ---- fused lm_head GEMM + online log-sum-exp NLL; block = 16 rows, 8 waves ----
__global__ __launch_bounds__(256)
void loss_kernel(const __bf16* __restrict__ X, const __bf16* __restrict__ LMH,
                 const int* __restrict__ target, float* __restrict__ out) {
  __shared__ __bf16 xls[16 * D_];             // 32 KB of normed rows
  __shared__ float red[8][16][3];
  int r0 = blockIdx.x << 4;
  int tid = threadIdx.x;
  {   // rows r0..r0+15 are contiguous: bulk copy as uint4
    const uint4* src = (const uint4*)(X + (size_t)r0 * D_);
    uint4* dst = (uint4*)xls;
    for (int i = tid; i < (16 * D_) / 8; i += 256) dst[i] = src[i];
  }
  __syncthreads();
  int wave = tid >> 5, lane = tid & 31;
  int n = lane & 15, rb = (lane >> 4) << 3;
  float m[8], l[8], tg[8];
  int lbl[8];
#pragma unroll
  for (int r = 0; r < 8; ++r) {
    m[r] = -1e30f; l[r] = 0.f; tg[r] = -1e30f;
    lbl[r] = target[r0 + rb + r];
  }
  for (int j = wave; j < V_ / 16; j += 8) {
    const __bf16* wrow = LMH + (size_t)(j << 4) * D_;
    v8f acc = {};
    for (int kc = 0; kc < D_; kc += 32) {
      acc = wmma_bf16(frag_a_bf(xls + kc, D_), frag_b_wt_bf(wrow + kc, D_), acc);
    }
    int col = (j << 4) + n;
#pragma unroll
    for (int r = 0; r < 8; ++r) {
      float v = acc[r];
      float mn = fmaxf(m[r], v);
      l[r] = l[r] * __expf(m[r] - mn) + __expf(v - mn);
      m[r] = mn;
      if (col == lbl[r]) tg[r] = v;
    }
  }
  for (int d = 1; d < 16; d <<= 1) {
#pragma unroll
    for (int r = 0; r < 8; ++r) {
      float mo = __shfl_xor(m[r], d, 32);
      float lo = __shfl_xor(l[r], d, 32);
      float to = __shfl_xor(tg[r], d, 32);
      float mn = fmaxf(m[r], mo);
      l[r] = l[r] * __expf(m[r] - mn) + lo * __expf(mo - mn);
      m[r] = mn;
      tg[r] = fmaxf(tg[r], to);
    }
  }
  if (n == 0) {
#pragma unroll
    for (int r = 0; r < 8; ++r) {
      red[wave][rb + r][0] = m[r];
      red[wave][rb + r][1] = l[r];
      red[wave][rb + r][2] = tg[r];
    }
  }
  __syncthreads();
  if (tid < 16) {
    float M = -1e30f, Ls = 0.f, T = -1e30f;
    for (int w = 0; w < 8; ++w) {
      float mo = red[w][tid][0], lo = red[w][tid][1], to = red[w][tid][2];
      float mn = fmaxf(M, mo);
      Ls = Ls * __expf(M - mn) + lo * __expf(mo - mn);
      M = mn;
      T = fmaxf(T, to);
    }
    int lab = target[r0 + tid];
    if (lab != IGNORE_IDX) atomicAdd(out, (M + __logf(Ls)) - T);
  }
}

// ---- host orchestration ----
extern "C" void kernel_launch(void* const* d_in, const int* /*in_sizes*/, int /*n_in*/,
                              void* d_out, int /*out_size*/, void* d_ws, size_t /*ws_size*/,
                              hipStream_t stream) {
  const int*   toks   = (const int*)d_in[0];
  const int*   target = (const int*)d_in[1];
  const float* embed  = (const float*)d_in[2];
  const float* Wq = (const float*)d_in[3];
  const float* Wk = (const float*)d_in[4];
  const float* Wv = (const float*)d_in[5];
  const float* Wo = (const float*)d_in[6];
  const float* ln1 = (const float*)d_in[7];
  const float* ln2 = (const float*)d_in[8];
  const float* gate_w = (const float*)d_in[9];
  const float* gate_projs = (const float*)d_in[10];
  const float* up_projs = (const float*)d_in[11];
  const float* down_projs = (const float*)d_in[12];
  const float* sg = (const float*)d_in[13];
  const float* su = (const float*)d_in[14];
  const float* sd = (const float*)d_in[15];
  const float* final_ln = (const float*)d_in[16];
  const float* lm_head = (const float*)d_in[17];
  float* out = (float*)d_out;

  // workspace carve-up (256B aligned)
  char* base = (char*)d_ws;
  auto alloc = [&](size_t bytes) -> char* {
    char* p = base;
    base += (bytes + 255) & ~(size_t)255;
    return p;
  };
  const size_t SD = (size_t)S_ * D_;
  const size_t SEI = (size_t)S_ * E_ * I_;
  const size_t SSI = (size_t)S_ * SI_;
  float*  h    = (float*)alloc(SD * 4);
  float*  q    = (float*)alloc(SD * 4);
  float*  k    = (float*)alloc(SD * 4);
  float*  v    = (float*)alloc(SD * 4);
  float*  g    = (float*)alloc(SEI * 4);       // later aliased as bf16 lm_head
  float*  u    = (float*)alloc(SEI * 4);
  float*  xg   = (float*)alloc(SSI * 4);
  float*  xu   = (float*)alloc(SSI * 4);
  float*  comb = (float*)alloc((size_t)S_ * E_ * 4);
  __bf16* xnb  = (__bf16*)alloc(SD * 2);
  __bf16* qb   = (__bf16*)alloc(SD * 2);
  __bf16* kb   = (__bf16*)alloc(SD * 2);
  __bf16* vb   = (__bf16*)alloc(SD * 2);       // transposed [D][S]
  __bf16* aob  = (__bf16*)alloc(SD * 2);
  __bf16* gb   = (__bf16*)alloc(SEI * 2);
  __bf16* xgb  = (__bf16*)alloc(SSI * 2);
  __bf16* wb   = (__bf16*)alloc((size_t)D_ * D_ * 2);   // layer-weight scratch
  __bf16* lmb  = (__bf16*)g;                   // V*D bf16 (64MB) over dead g/u (2 x 32MB)

  dim3 gD(D_ / 64, S_ / 32), gI(I_ / 64, S_ / 32), gSI(SI_ / 64, S_ / 32);
  const int CVT_DD = (D_ * D_) / 1024, CVT_ID = (I_ * D_) / 1024;
  const int CVT_DSI = (D_ * SI_) / 1024;

  embed_kernel<<<(S_ * D_) / 256, 256, 0, stream>>>(toks, embed, h);

  for (int l = 0; l < L_; ++l) {
    rmsnorm_kernel<<<S_, 256, 0, stream>>>(h, ln1 + (size_t)l * D_, xnb);
    cvt_bf16_kernel<<<CVT_DD, 256, 0, stream>>>(Wq + (size_t)l * D_ * D_, wb);
    gemm_bf<false><<<gD, 32, 0, stream>>>(xnb, wb, q, D_, D_, D_, D_);
    cvt_bf16_kernel<<<CVT_DD, 256, 0, stream>>>(Wk + (size_t)l * D_ * D_, wb);
    gemm_bf<false><<<gD, 32, 0, stream>>>(xnb, wb, k, D_, D_, D_, D_);
    cvt_bf16_kernel<<<CVT_DD, 256, 0, stream>>>(Wv + (size_t)l * D_ * D_, wb);
    gemm_bf<false><<<gD, 32, 0, stream>>>(xnb, wb, v, D_, D_, D_, D_);
    rope_kernel<<<(S_ * H_ * (DH_ / 2)) / 256, 256, 0, stream>>>(q, k, qb, kb);
    cvt_vT_kernel<<<(S_ * D_) / 256, 256, 0, stream>>>(v, vb);
    attn_kernel<<<dim3(S_ / 16, H_), 32, 0, stream>>>(qb, kb, vb, aob);
    cvt_bf16_kernel<<<CVT_DD, 256, 0, stream>>>(Wo + (size_t)l * D_ * D_, wb);
    gemm_bf<true><<<gD, 32, 0, stream>>>(aob, wb, h, D_, D_, D_, D_);

    rmsnorm_kernel<<<S_, 256, 0, stream>>>(h, ln2 + (size_t)l * D_, xnb);
    gate_kernel<<<S_, 32, 0, stream>>>(xnb, gate_w + (size_t)l * E_ * D_, comb);
    for (int e = 0; e < E_; ++e) {
      cvt_bf16_kernel<<<CVT_ID, 256, 0, stream>>>(
          gate_projs + (size_t)(l * E_ + e) * I_ * D_, wb);
      gemm_bf<false><<<gI, 32, 0, stream>>>(xnb, wb, g + e * I_, D_, D_, D_, E_ * I_);
      cvt_bf16_kernel<<<CVT_ID, 256, 0, stream>>>(
          up_projs + (size_t)(l * E_ + e) * I_ * D_, wb);
      gemm_bf<false><<<gI, 32, 0, stream>>>(xnb, wb, u + e * I_, D_, D_, D_, E_ * I_);
    }
    moe_act_kernel<<<(S_ * E_ * I_) / 256, 256, 0, stream>>>(g, u, comb, gb);
    for (int e = 0; e < E_; ++e) {
      cvt_bf16_kernel<<<CVT_ID, 256, 0, stream>>>(
          down_projs + (size_t)(l * E_ + e) * D_ * I_, wb);
      gemm_bf<true><<<gD, 32, 0, stream>>>(gb + e * I_, wb, h, I_, E_ * I_, I_, D_);
    }
    cvt_bf16_kernel<<<CVT_DSI, 256, 0, stream>>>(sg + (size_t)l * SI_ * D_, wb);
    gemm_bf<false><<<gSI, 32, 0, stream>>>(xnb, wb, xg, D_, D_, D_, SI_);
    cvt_bf16_kernel<<<CVT_DSI, 256, 0, stream>>>(su + (size_t)l * SI_ * D_, wb);
    gemm_bf<false><<<gSI, 32, 0, stream>>>(xnb, wb, xu, D_, D_, D_, SI_);
    silu_mul_kernel<<<(S_ * SI_) / 256, 256, 0, stream>>>(xg, xu, xgb);
    cvt_bf16_kernel<<<CVT_DSI, 256, 0, stream>>>(sd + (size_t)l * D_ * SI_, wb);
    gemm_bf<true><<<gD, 32, 0, stream>>>(xgb, wb, h, SI_, SI_, SI_, D_);
  }

  rmsnorm_kernel<<<S_, 256, 0, stream>>>(h, final_ln, xnb);
  cvt_bf16_kernel<<<((size_t)V_ * D_) / 1024, 256, 0, stream>>>(lm_head, lmb);
  zero_kernel<<<1, 1, 0, stream>>>(out);
  loss_kernel<<<S_ / 16, 256, 0, stream>>>(xnb, lmb, target, out);
}
